// BatchedGATWrapper_35089882808457
// MI455X (gfx1250) — matrix operand order repaired and checked by reading, not compiled, not run on hardware
//
#include <hip/hip_runtime.h>

#define DEVINL __device__ __forceinline__

typedef __attribute__((ext_vector_type(16))) __bf16 v16bf;
typedef __attribute__((ext_vector_type(8)))  float  v8f;
typedef __attribute__((ext_vector_type(2)))  float  v2f;
typedef __attribute__((ext_vector_type(2)))  __bf16 v2bf;

union FragBF { v16bf v; unsigned int u[8]; uint4 q[2]; };

constexpr int Bn = 4, Nn = 2048, Dn = 512, Hn = 8, Cn = 64, HC = 512;
constexpr int NW = Nn / 32;   // mask words per target row

// packed float2 -> bf16x2 (v_cvt_pk_bf16_f32)
DEVINL unsigned int pack2bf(float lo, float hi) {
  union { v2bf b; unsigned int u; } cv;
  cv.b = __builtin_convertvector((v2f){lo, hi}, v2bf);
  return cv.u;
}

// 16-bit A-matrix 16x32 layout (ISA 7.12.2): lane half selects K base 0/8,
// VGPRs 0-3 -> K pairs in [base, base+8), VGPRs 4-7 -> [16+base, 16+base+8)
DEVINL int kmapA(int v, int lane) {
  return ((v & 4) << 2) + ((lane >> 4) << 3) + ((v & 3) << 1);
}

// ---------------------------------------------------------------------------
// Streaming pre-conversions (run once per launch, memory-bound)
// ---------------------------------------------------------------------------
__global__ __launch_bounds__(256)
void convert_feat(const float* __restrict__ src, unsigned int* __restrict__ dst) {
  const size_t t = (size_t)blockIdx.x * 256 + threadIdx.x;
  const float4 f = ((const float4*)src)[t];
  dst[2 * t]     = pack2bf(f.x, f.y);
  dst[2 * t + 1] = pack2bf(f.z, f.w);
}

// W [k][col] fp32 -> WT [col][k] bf16 (K-contiguous for B fragments)
__global__ __launch_bounds__(256)
void convert_W(const float* __restrict__ W, unsigned int* __restrict__ WT) {
  const int col = blockIdx.x;
  const int k   = threadIdx.x << 1;
  WT[col * (Dn / 2) + threadIdx.x] =
      pack2bf(W[(size_t)k * HC + col], W[(size_t)(k + 1) * HC + col]);
}

// adj int32 [b][i][j] -> bitmask [b][j][i/32].
// Stages the tile via CDNA5 async global->LDS copies (ASYNCcnt), then
// transposes with wave32 ballots. 64 MB adjacency is read exactly once.
__global__ __launch_bounds__(256)
void pack_mask(const int* __restrict__ adj, unsigned int* __restrict__ maskP) {
  __shared__ int tile[32 * 33];                       // pad to kill bank conflicts
  const int b  = blockIdx.z;
  const int i0 = blockIdx.y << 5;
  const int j0 = blockIdx.x << 5;
  const int tid = threadIdx.y * 32 + threadIdx.x;
#pragma unroll
  for (int e = tid; e < 1024; e += 256) {
    const int ii = e >> 5, jj = e & 31;
    const unsigned lds_off = (unsigned)((ii * 33 + jj) * 4);  // tile at LDS offset 0
    const int* gp = adj + (size_t)(b * Nn + i0 + ii) * Nn + j0 + jj;
    asm volatile("global_load_async_to_lds_b32 %0, %1, off"
                 :: "v"(lds_off), "v"(gp) : "memory");
  }
  asm volatile("s_wait_asynccnt 0x0" ::: "memory");   // drain this wave's copies
  __syncthreads();
  const int lane = threadIdx.x;
#pragma unroll
  for (int q = 0; q < 4; ++q) {
    const int jj = threadIdx.y + (q << 3);
    const unsigned long long bal = __ballot(tile[lane * 33 + jj] != 0);
    if (lane == 0)
      maskP[(size_t)(b * Nn + j0 + jj) * NW + (i0 >> 5)] = (unsigned int)bal;
  }
}

// ---------------------------------------------------------------------------
// Kernel 1: x = feat @ W via bf16 WMMA. M-tile = 32 rows per wave so each
// B fragment (W tile) feeds 2 WMMAs; all fragments are 2x B128 loads.
// Emits xT [b][h][c][n] bf16 and aS/aD [b][h][n] f32 (from fp32 accumulators).
// ---------------------------------------------------------------------------
__global__ __launch_bounds__(256)
void gat_transform(const unsigned short* __restrict__ featBF,
                   const unsigned short* __restrict__ WT,
                   const float* __restrict__ att_src, const float* __restrict__ att_dst,
                   unsigned short* __restrict__ xT,
                   float* __restrict__ aS, float* __restrict__ aD) {
  const int lane = threadIdx.x;
  const int task = blockIdx.x * 8 + threadIdx.y;   // 2048 wave-tasks
  const int h    = task & 7;
  const int rt   = task >> 3;                      // 256 x 32-row tiles
  const int b    = rt >> 6;
  const int nbase = (rt & 63) << 5;
  const int half = lane >> 4;
  const int lj   = lane & 15;

  const unsigned short* frow0 = featBF + (size_t)(b * Nn + nbase + lj) * Dn;
  const unsigned short* frow1 = frow0 + (size_t)16 * Dn;

  v8f acc[2][4] = {};
  for (int kk = 0; kk < Dn; kk += 32) {
    FragBF a0, a1;                                  // A: K pairs contiguous
    a0.q[0] = *(const uint4*)(frow0 + kk + (half << 3));
    a0.q[1] = *(const uint4*)(frow0 + kk + (half << 3) + 16);
    a1.q[0] = *(const uint4*)(frow1 + kk + (half << 3));
    a1.q[1] = *(const uint4*)(frow1 + kk + (half << 3) + 16);
#pragma unroll
    for (int w = 0; w < 4; ++w) {
      const int col = h * Cn + w * 16 + lj;
      const unsigned short* wp = WT + (size_t)col * Dn + kk + (half << 3);
      FragBF bb;                                    // B: reused by both M-tiles
      bb.q[0] = *(const uint4*)(wp);
      bb.q[1] = *(const uint4*)(wp + 16);
      acc[0][w] = __builtin_amdgcn_wmma_f32_16x16x32_bf16(
          false, a0.v, false, bb.v, (short)0, acc[0][w], false, false);
      acc[1][w] = __builtin_amdgcn_wmma_f32_16x16x32_bf16(
          false, a1.v, false, bb.v, (short)0, acc[1][w], false, false);
    }
  }

  unsigned int* xc32 = (unsigned int*)(xT + (((size_t)(b * Hn + h) * Cn) << 11));
  float asv[4], adv[4];
#pragma unroll
  for (int w = 0; w < 4; ++w) {
    asv[w] = att_src[h * Cn + w * 16 + lj];
    adv[w] = att_dst[h * Cn + w * 16 + lj];
  }

#pragma unroll
  for (int t = 0; t < 2; ++t) {
    const int nb = nbase + t * 16;
    // store xT (bf16, packed pair stores: consecutive r = consecutive n)
#pragma unroll
    for (int w = 0; w < 4; ++w)
#pragma unroll
      for (int r = 0; r < 8; r += 2)
        xc32[((((size_t)(w * 16 + lj)) << 11) + nb + half * 8 + r) >> 1] =
            pack2bf(acc[t][w][r], acc[t][w][r + 1]);

#pragma unroll
    for (int r = 0; r < 8; ++r) {
      float ps = 0.f, pd = 0.f;
#pragma unroll
      for (int w = 0; w < 4; ++w) { ps += acc[t][w][r] * asv[w]; pd += acc[t][w][r] * adv[w]; }
#pragma unroll
      for (int off = 1; off < 16; off <<= 1) {
        ps += __shfl_xor(ps, off, 32);
        pd += __shfl_xor(pd, off, 32);
      }
      if (lj == r) {
        const int row = nb + half * 8 + r;
        aS[(b * Hn + h) * Nn + row] = ps;
        aD[(b * Hn + h) * Nn + row] = pd;
      }
    }
  }
}

// ---------------------------------------------------------------------------
// Kernel 2: fused masked-softmax + aggregation (flash-style; alpha never hits
// memory). One block = (b, 32-target tile), 8 waves = 8 heads; each wave runs
// TWO 16-target tiles so every x fragment and a_src block feeds 2 alpha tiles
// (8 WMMAs per step on 8 B128 loads). Mask = packed bitmask, 1 u32 / 32 srcs.
// ---------------------------------------------------------------------------
__global__ __launch_bounds__(256)
void gat_aggregate(const unsigned int* __restrict__ maskP,
                   const unsigned short* __restrict__ xT,
                   const float* __restrict__ aS, const float* __restrict__ aD,
                   const float* __restrict__ bias, float* __restrict__ out) {
  const int lane = threadIdx.x, h = threadIdx.y;
  const int b = blockIdx.y;
  const int jbase = blockIdx.x << 5;               // 32 targets per block
  const int half = lane >> 4;
  const int lj   = lane & 15;

  const float* asb = aS + (b * Hn + h) * Nn;
  const unsigned short* xb = xT + ((size_t)(b * Hn + h) << 17);

  int   jglob[2];
  float ad[2], m[2], lsum[2];
  const unsigned int* mrow[2];
#pragma unroll
  for (int t = 0; t < 2; ++t) {
    jglob[t] = jbase + t * 16 + lj;
    ad[t]    = aD[(b * Hn + h) * Nn + jglob[t]];
    mrow[t]  = maskP + (size_t)(b * Nn + jglob[t]) * NW;
    m[t] = -1e9f; lsum[t] = 0.f;
  }
  v8f acc[2][4] = {};

  for (int kb = 0; kb < Nn; kb += 32) {
    unsigned int mW[2];
#pragma unroll
    for (int t = 0; t < 2; ++t) {
      mW[t] = mrow[t][kb >> 5];
      const int d = jglob[t] - kb;                 // self-loop bit
      if (0 <= d && d < 32) mW[t] |= (1u << d);
    }
    if (__ballot((mW[0] | mW[1]) != 0) == 0ull) continue;  // wave-uniform skip

    // a_src for this lane's 16 source slots (j-independent: shared by both tiles)
    const float* ap = asb + kb + (half << 3);
    const float4 s0 = *(const float4*)(ap);
    const float4 s1 = *(const float4*)(ap + 4);
    const float4 s2 = *(const float4*)(ap + 16);
    const float4 s3 = *(const float4*)(ap + 20);
    const float asf[16] = {s0.x, s0.y, s0.z, s0.w, s1.x, s1.y, s1.z, s1.w,
                           s2.x, s2.y, s2.z, s2.w, s3.x, s3.y, s3.z, s3.w};

    FragBF afr[2];
    float scale[2];
#pragma unroll
    for (int t = 0; t < 2; ++t) {
      float lg[16];
      float mblk = -1e9f;
#pragma unroll
      for (int v = 0; v < 8; ++v) {
        const unsigned int bits = mW[t] >> kmapA(v, lane);
        float x0 = asf[2 * v] + ad[t];
        float x1 = asf[2 * v + 1] + ad[t];
        x0 = x0 > 0.f ? x0 : 0.2f * x0;            // leaky_relu(0.2)
        x1 = x1 > 0.f ? x1 : 0.2f * x1;
        x0 = (bits & 1u) ? x0 : -1e9f;
        x1 = (bits & 2u) ? x1 : -1e9f;
        lg[2 * v] = x0; lg[2 * v + 1] = x1;
        mblk = fmaxf(mblk, fmaxf(x0, x1));
      }
      mblk = fmaxf(mblk, __shfl_xor(mblk, 16, 32));  // lane pair -> all 32 i's
      const float mnew = fmaxf(m[t], mblk);
      scale[t] = __expf(m[t] - mnew);
      m[t] = mnew;

      float rs = 0.f;
#pragma unroll
      for (int v = 0; v < 8; ++v) {
        const float p0 = lg[2 * v]     > -1e8f ? __expf(lg[2 * v]     - mnew) : 0.f;
        const float p1 = lg[2 * v + 1] > -1e8f ? __expf(lg[2 * v + 1] - mnew) : 0.f;
        rs += p0 + p1;
        afr[t].u[v] = pack2bf(p0, p1);
      }
      rs += __shfl_xor(rs, 16, 32);
      lsum[t] = lsum[t] * scale[t] + rs;
    }

    if (__ballot(fminf(scale[0], scale[1]) < 0.9999999f)) {  // skip when max stable
#pragma unroll
      for (int r = 0; r < 8; ++r) {
        const float sA = __shfl(scale[0], (half << 3) + r, 32);
        const float sB = __shfl(scale[1], (half << 3) + r, 32);
#pragma unroll
        for (int w = 0; w < 4; ++w) { acc[0][w][r] *= sA; acc[1][w][r] *= sB; }
      }
    }

#pragma unroll
    for (int w = 0; w < 4; ++w) {
      const unsigned short* xp =
          xb + (((size_t)(w * 16 + lj)) << 11) + kb + (half << 3);
      FragBF bf;                                   // B: 2x B128, reused by 2 tiles
      bf.q[0] = *(const uint4*)(xp);
      bf.q[1] = *(const uint4*)(xp + 16);
      __builtin_prefetch(xp + 32, 0, 0);           // next source block
      acc[0][w] = __builtin_amdgcn_wmma_f32_16x16x32_bf16(
          false, afr[0].v, false, bf.v, (short)0, acc[0][w], false, false);
      acc[1][w] = __builtin_amdgcn_wmma_f32_16x16x32_bf16(
          false, afr[1].v, false, bf.v, (short)0, acc[1][w], false, false);
    }
  }

  // epilogue: normalize, bias, ELU
#pragma unroll
  for (int t = 0; t < 2; ++t)
#pragma unroll
    for (int r = 0; r < 8; ++r) {
      const float linv = 1.f / __shfl(lsum[t], (half << 3) + r, 32);
      const int j = jbase + t * 16 + (half << 3) + r;
#pragma unroll
      for (int w = 0; w < 4; ++w) {
        const int c = h * Cn + w * 16 + lj;
        float val = acc[t][w][r] * linv + bias[c];
        val = val > 0.f ? val : __expf(val) - 1.f;
        out[(size_t)(b * Nn + j) * HC + c] = val;
      }
    }
}

extern "C" void kernel_launch(void* const* d_in, const int* in_sizes, int n_in,
                              void* d_out, int out_size, void* d_ws, size_t ws_size,
                              hipStream_t stream) {
  (void)in_sizes; (void)n_in; (void)out_size; (void)ws_size;
  const float* feat    = (const float*)d_in[0];
  const int*   adj     = (const int*)d_in[1];
  const float* W       = (const float*)d_in[2];
  const float* att_src = (const float*)d_in[3];
  const float* att_dst = (const float*)d_in[4];
  const float* bias    = (const float*)d_in[5];
  float* out = (float*)d_out;

  // workspace layout (all rewritten every call; ~19 MB total)
  size_t off = 0;
  unsigned short* xT     = (unsigned short*)((char*)d_ws + off); off += (size_t)Bn * Hn * Cn * Nn * 2; // 8 MB
  unsigned short* featBF = (unsigned short*)((char*)d_ws + off); off += (size_t)Bn * Nn * Dn * 2;      // 8 MB
  unsigned short* WT     = (unsigned short*)((char*)d_ws + off); off += (size_t)Dn * HC * 2;           // 512 KB
  unsigned int*   maskP  = (unsigned int*)  ((char*)d_ws + off); off += (size_t)Bn * Nn * NW * 4;      // 2 MB
  float*          aS     = (float*)         ((char*)d_ws + off); off += (size_t)Bn * Hn * Nn * 4;      // 256 KB
  float*          aD     = (float*)         ((char*)d_ws + off);

  const dim3 blk(32, 8);
  convert_feat<<<(Bn * Nn * Dn / 4) / 256, 256, 0, stream>>>(feat, (unsigned int*)featBF);
  convert_W<<<HC, 256, 0, stream>>>(W, (unsigned int*)WT);
  pack_mask<<<dim3(Nn / 32, Nn / 32, Bn), blk, 0, stream>>>(adj, maskP);
  gat_transform<<<Bn * (Nn / 32) * Hn / 8, blk, 0, stream>>>(featBF, WT, att_src, att_dst, xT, aS, aD);
  gat_aggregate<<<dim3(Nn / 32, Bn), blk, 0, stream>>>(maskP, xT, aS, aD, bias, out);
}
